// StockPredictor_46462956208690
// MI455X (gfx1250) — compile-verified, matrix operand order checked by reading, not run on hardware
//
#include <hip/hip_runtime.h>

#define HDIM 50
#define GDIM 200     // 4*H
#define NPAD 208     // 13 tiles of 16
#define NT   13
#define KP   64      // padded K for H=50
#define TSEQ 512
#define BATCH 2048

typedef __attribute__((ext_vector_type(16))) _Float16 v16h;
typedef __attribute__((ext_vector_type(8)))  _Float16 v8h;
typedef __attribute__((ext_vector_type(8)))  float    v8f;
typedef __attribute__((ext_vector_type(4)))  float    v4f;
typedef __attribute__((ext_vector_type(2)))  float    v2f;

#if __has_builtin(__builtin_amdgcn_tanhf)
  #define TANHF(x) __builtin_amdgcn_tanhf(x)
#elif __has_builtin(__builtin_amdgcn_tanh_f32)
  #define TANHF(x) __builtin_amdgcn_tanh_f32(x)
#else
  #define TANHF(x) (1.0f - 2.0f / (__expf(2.0f * (x)) + 1.0f))
#endif

__device__ __forceinline__ float sigm(float v) { return 0.5f * TANHF(0.5f * v) + 0.5f; }

__device__ __forceinline__ v8f wmma16(v16h a, v16h b, v8f c) {
  return __builtin_amdgcn_wmma_f32_16x16x32_f16(false, a, false, b, (short)0, c, false, false);
}

// CDNA5 16-bit operand fragment: lane group g reads K-chunks [base..base+7] and
// [base+16..base+23] as two 16B LDS loads. p points at (row, kbase + 8*g).
__device__ __forceinline__ v16h ld_frag(const _Float16* p) {
  v16h r;
  *(v8h*)&r         = *(const v8h*)p;
  *(((v8h*)&r) + 1) = *(const v8h*)(p + 16);
  return r;
}

// one M-pair of LSTM cell update: gates from sG ([N][16] f32), c in regs, h -> sH (f16 [16][64])
#define EW_BODY(PID, CREG, QIDX, SH) do {                                   \
    const int j_ = (PID) >> 3;                                              \
    const int m_ = ((PID) & 7) * 2;                                         \
    const float* gb_ = sG + j_*16 + m_;                                     \
    v2f gi2 = *(const v2f*)(gb_);                                           \
    v2f gf2 = *(const v2f*)(gb_ + HDIM*16);                                 \
    v2f gg2 = *(const v2f*)(gb_ + 2*HDIM*16);                               \
    v2f go2 = *(const v2f*)(gb_ + 3*HDIM*16);                               \
    _Pragma("unroll")                                                       \
    for (int u_ = 0; u_ < 2; ++u_) {                                        \
      float i_ = sigm(gi2[u_]);                                             \
      float f_ = sigm(gf2[u_]);                                             \
      float g_ = TANHF(gg2[u_]);                                            \
      float o_ = sigm(go2[u_]);                                             \
      float c_ = f_ * CREG[QIDX][u_] + i_ * g_;                             \
      CREG[QIDX][u_] = c_;                                                  \
      SH[(m_ + u_)*KP + j_] = (_Float16)(o_ * TANHF(c_));                   \
    }                                                                       \
  } while (0)

extern "C" __global__ __launch_bounds__(32, 1)
void lstm2_wmma(const float* __restrict__ x,
                const float* __restrict__ Wih0, const float* __restrict__ Whh0,
                const float* __restrict__ bih0, const float* __restrict__ bhh0,
                const float* __restrict__ Wih1, const float* __restrict__ Whh1,
                const float* __restrict__ bih1, const float* __restrict__ bhh1,
                const float* __restrict__ fcw,  const float* __restrict__ fcb,
                float* __restrict__ out)
{
  extern __shared__ char smem[];
  _Float16* sWihA = (_Float16*)smem;                   // [208][32] (K=0..2 = W, K=3 = bias0)
  _Float16* sWhhA = sWihA + NPAD*32;                   // [208][64]
  _Float16* sWihB = sWhhA + NPAD*64;                   // [208][64] (K=50 = bias1)
  _Float16* sWhhB = sWihB + NPAD*64;                   // [208][64]
  _Float16* sHA   = sWhhB + NPAD*64;                   // [16][64] h layer0 (col 50 == 1.0)
  _Float16* sHB   = sHA + 16*KP;                       // [16][64] h layer1
  float*    sG    = (float*)(sHB + 16*KP);             // [208][16] gate pre-acts

  const int tid = threadIdx.x;
  const int lm  = tid & 15;
  const int gh  = tid >> 4;
  const int m0  = blockIdx.x * 16;

  // ---- stage weights fp32 -> fp16, padded; biases folded into K-columns ----
  for (int i = tid; i < NPAD*32; i += 32) {
    int n = i >> 5, k = i & 31;
    float v = 0.0f;
    if (n < GDIM) {
      if (k < 3)       v = Wih0[n*3 + k];
      else if (k == 3) v = bih0[n] + bhh0[n];   // bias via ax[K=3] == 1
    }
    sWihA[i] = (_Float16)v;
  }
  for (int i = tid; i < NPAD*64; i += 32) {
    int n = i >> 6, k = i & 63;
    bool ok = (n < GDIM) && (k < HDIM);
    sWhhA[i] = (_Float16)(ok ? Whh0[n*HDIM + k] : 0.0f);
    float vb = 0.0f;
    if (n < GDIM) {
      if (k < HDIM)       vb = Wih1[n*HDIM + k];
      else if (k == HDIM) vb = bih1[n] + bhh1[n];   // bias via sHA[m][50] == 1
    }
    sWihB[i] = (_Float16)vb;
    sWhhB[i] = (_Float16)(ok ? Whh1[n*HDIM + k] : 0.0f);
  }
  for (int i = tid; i < 16*KP; i += 32) {
    int k = i & 63;
    sHA[i] = (_Float16)((k == HDIM) ? 1.0f : 0.0f);  // constant-1 bias column
    sHB[i] = (_Float16)0.0f;
  }
  __syncthreads();

  // cell states in registers (each lane owns 13 M-pairs per layer)
  float cA[NT][2], cB[NT][2];
  #pragma unroll
  for (int q = 0; q < NT; ++q) { cA[q][0] = cA[q][1] = 0.0f; cB[q][0] = cB[q][1] = 0.0f; }

  v8f zero8;
  #pragma unroll
  for (int e = 0; e < 8; ++e) zero8[e] = 0.0f;

  const float* px = x + ((size_t)(m0 + lm) * TSEQ) * 3;
  const float  one = (gh == 0) ? 1.0f : 0.0f;

  // pre-load x_0 (pipelined one step ahead thereafter)
  float xl0 = 0.0f, xl1 = 0.0f, xl2 = 0.0f;
  if (gh == 0) { xl0 = px[0]; xl1 = px[1]; xl2 = px[2]; }

  const _Float16* wx = sWihA + lm*32 + 8*gh;   // + n*512 per tile
  const _Float16* wh = sWhhA + lm*64 + 8*gh;   // + n*1024
  const _Float16* wi = sWihB + lm*64 + 8*gh;
  const _Float16* wr = sWhhB + lm*64 + 8*gh;
  const _Float16* pa = sHA + lm*KP + 8*gh;
  const _Float16* pb = sHB + lm*KP + 8*gh;

  for (int t = 0; t < TSEQ; ++t) {
    // x_t fragment: K=0..2 = x, K=3 = 1 (bias); only gh==0 lanes carry K<8
    v16h ax;
    #pragma unroll
    for (int e = 0; e < 16; ++e) ax[e] = (_Float16)0.0f;
    ax[0] = (_Float16)xl0; ax[1] = (_Float16)xl1; ax[2] = (_Float16)xl2;
    ax[3] = (_Float16)one;
    // prefetch x_{t+1} (consumed next iteration -> latency hidden)
    if (gh == 0 && t + 1 < TSEQ) {
      const float* p = px + (t + 1) * 3;
      xl0 = p[0]; xl1 = p[1]; xl2 = p[2];
    }

    // previous-step h fragments (capture before overwriting)
    v16h haf0 = ld_frag(pa);
    v16h haf1 = ld_frag(pa + 32);
    v16h hbf0 = ld_frag(pb);
    v16h hbf1 = ld_frag(pb + 32);

    // ---- layer 0 gates (pipelined B loads) -> sG ----
    {
      v16h bx = ld_frag(wx), b0 = ld_frag(wh), b1 = ld_frag(wh + 32);
      #pragma unroll
      for (int n = 0; n < NT; ++n) {
        const int nn = (n + 1 < NT) ? (n + 1) : n;
        v16h pbx = ld_frag(wx + nn*512);
        v16h pb0 = ld_frag(wh + nn*1024);
        v16h pb1 = ld_frag(wh + nn*1024 + 32);
        v8f acc = wmma16(ax,   bx, zero8);
        acc     = wmma16(haf0, b0, acc);
        acc     = wmma16(haf1, b1, acc);
        float* gp = sG + (n*16 + lm)*16 + 8*gh;   // [N][M]
        *(v4f*)gp       = *(v4f*)&acc;
        *(v4f*)(gp + 4) = *(((v4f*)&acc) + 1);
        bx = pbx; b0 = pb0; b1 = pb1;
      }
    }

    // ---- layer 1 recurrent partials (independent of layer-0 output):
    //      issued now so they drain while the layer-0 elementwise VALU runs ----
    v8f pacc[NT];
    {
      v16h b2 = ld_frag(wr), b3 = ld_frag(wr + 32);
      #pragma unroll
      for (int n = 0; n < NT; ++n) {
        const int nn = (n + 1 < NT) ? (n + 1) : n;
        v16h p2 = ld_frag(wr + nn*1024);
        v16h p3 = ld_frag(wr + nn*1024 + 32);
        v8f a   = wmma16(hbf0, b2, zero8);
        pacc[n] = wmma16(hbf1, b3, a);
        b2 = p2; b3 = p3;
      }
    }

    // ---- layer 0 elementwise (12 full iters + tail; co-executes with WMMA drain) ----
    #pragma unroll
    for (int q = 0; q < 12; ++q) { int pid = q*32 + tid; EW_BODY(pid, cA, q, sHA); }
    if (tid < 16) { int pid = 384 + tid; EW_BODY(pid, cA, 12, sHA); }

    // new layer-0 h is layer-1 input (includes bias column at K=50)
    v16h naf0 = ld_frag(pa);
    v16h naf1 = ld_frag(pa + 32);

    // ---- layer 1 input gates: add into recurrent partials -> sG ----
    {
      v16h b0 = ld_frag(wi), b1 = ld_frag(wi + 32);
      #pragma unroll
      for (int n = 0; n < NT; ++n) {
        const int nn = (n + 1 < NT) ? (n + 1) : n;
        v16h p0 = ld_frag(wi + nn*1024);
        v16h p1 = ld_frag(wi + nn*1024 + 32);
        v8f acc = wmma16(naf0, b0, pacc[n]);
        acc     = wmma16(naf1, b1, acc);
        float* gp = sG + (n*16 + lm)*16 + 8*gh;
        *(v4f*)gp       = *(v4f*)&acc;
        *(v4f*)(gp + 4) = *(((v4f*)&acc) + 1);
        b0 = p0; b1 = p1;
      }
    }

    // ---- layer 1 elementwise ----
    #pragma unroll
    for (int q = 0; q < 12; ++q) { int pid = q*32 + tid; EW_BODY(pid, cB, q, sHB); }
    if (tid < 16) { int pid = 384 + tid; EW_BODY(pid, cB, 12, sHB); }
  }

  // ---- final FC: out = hB_last @ fc_w^T + fc_b ----
  if (tid < 16) {
    #pragma unroll
    for (int o = 0; o < 3; ++o) {
      float acc = fcb[o];
      for (int j = 0; j < HDIM; ++j)
        acc += (float)sHB[tid*KP + j] * fcw[o*HDIM + j];
      out[(size_t)(m0 + tid)*3 + o] = acc;
    }
  }
}

extern "C" void kernel_launch(void* const* d_in, const int* in_sizes, int n_in,
                              void* d_out, int out_size, void* d_ws, size_t ws_size,
                              hipStream_t stream) {
  const float* x    = (const float*)d_in[0];
  const float* Wih0 = (const float*)d_in[1];
  const float* Whh0 = (const float*)d_in[2];
  const float* bih0 = (const float*)d_in[3];
  const float* bhh0 = (const float*)d_in[4];
  const float* Wih1 = (const float*)d_in[5];
  const float* Whh1 = (const float*)d_in[6];
  const float* bih1 = (const float*)d_in[7];
  const float* bhh1 = (const float*)d_in[8];
  const float* fcw  = (const float*)d_in[9];
  const float* fcb  = (const float*)d_in[10];
  float* out = (float*)d_out;

  // LDS: WihA 13312 + 3*26624 + h 4096 + G 13312 = 110592 B
  const size_t smem = 13312 + 3*26624 + 4096 + 13312;

  lstm2_wmma<<<dim3(BATCH/16), dim3(32), smem, stream>>>(
      x, Wih0, Whh0, bih0, bhh0, Wih1, Whh1, bih1, bhh1, fcw, fcb, out);
}
